// OriLinearGNN_59322088292863
// MI455X (gfx1250) — matrix operand order repaired and checked by reading, not compiled, not run on hardware
//
#include <hip/hip_runtime.h>
#include <hip/hip_bf16.h>

// ---- problem constants (match reference) ----
#define LNF   64
#define SDIM  8
#define TSTEP 4
#define NET   10
#define MUF   0.9f
#define KPAD  160      // 138 -> 160 (5 chunks of K=32)
#define NPAD  80       // 64 (A cols) + 8 (rou cols) -> 80 (5 tiles of N=16)

typedef __attribute__((ext_vector_type(16))) __bf16 v16bf;
typedef __attribute__((ext_vector_type(8)))  float  v8f;

typedef union { uint4 q[2]; v16bf v; } frag_u;

// hardware V_TANH_F32 when available (gfx1250 trans op), libm fallback
#if __has_builtin(__builtin_amdgcn_tanhf)
#define DTANH(x) __builtin_amdgcn_tanhf(x)
#elif __has_builtin(__builtin_amdgcn_tanh_f32)
#define DTANH(x) __builtin_amdgcn_tanh_f32(x)
#else
#define DTANH(x) tanhf(x)
#endif

static __device__ __forceinline__ unsigned short f2bf(float f) {
  unsigned int u = __float_as_uint(f);
  u += 0x7FFFu + ((u >> 16) & 1u);          // round-to-nearest-even
  return (unsigned short)(u >> 16);
}
static __device__ __forceinline__ unsigned int pack2bf(float a, float b) {
  return (unsigned int)f2bf(a) | ((unsigned int)f2bf(b) << 16);
}

// ---------- prep: bf16 weight block [NPAD][KPAD]; copy H_init -> Ha ----------
__global__ void k_prep(const float* __restrict__ Wxi, const float* __restrict__ Wrou,
                       const float* __restrict__ Hinit,
                       unsigned short* __restrict__ Wbig, float* __restrict__ Ha, int V) {
  const int stride = gridDim.x * blockDim.x;
  const int i0 = blockIdx.x * blockDim.x + threadIdx.x;
  for (int i = i0; i < NPAD * KPAD; i += stride) {
    int n = i / KPAD, k = i % KPAD;
    float v = 0.0f;
    if (n < 64)      { if (k < 138) v = Wxi[n * 138 + k]; }
    else if (n < 72) { if (k >= 64 && k < 128) v = Wrou[(n - 64) * 64 + (k - 64)]; }
    Wbig[i] = f2bf(v);
  }
  for (int i = i0; i < V * SDIM; i += stride) Ha[i] = Hinit[i];
}

__global__ void k_zero(float* __restrict__ p, int n) {
  for (int i = blockIdx.x * blockDim.x + threadIdx.x; i < n; i += gridDim.x * blockDim.x)
    p[i] = 0.0f;
}

// ---------- fused recurrence step: A/bias via WMMA, 8x8 matvec, atomic scatter ----
__global__ __launch_bounds__(128, 2) void k_step(
    const float* __restrict__ feat, const int* __restrict__ Xn,
    const int* __restrict__ Xe, const int* __restrict__ Et,
    const float* __restrict__ dg, const unsigned short* __restrict__ Wbig,
    const float* __restrict__ bxi, const float* __restrict__ brou,
    const float* __restrict__ Hc, float* __restrict__ Hn, int E, int V) {
  __shared__ unsigned short Wl[NPAD * KPAD];     // 25.0 KB  weights (bf16)
  __shared__ unsigned short Xl[4][16 * KPAD];    // 20.0 KB  per-wave 16-edge X tiles
  __shared__ float Al[4][16 * 64];               // 16.0 KB  scaled tanh(A)
  __shared__ float Bl[4][16 * 8];                //  2.0 KB  bias
  __shared__ float Sc[4][16];
  __shared__ int   Se[4][16];
  __shared__ int   Ne[4][16];

  const int tid  = threadIdx.x;
  const int wid  = tid >> 5;
  const int lane = tid & 31;

  // ---- stage weight block to LDS: async global->LDS b128 DMA when available ----
#if defined(__gfx1250__) && __has_builtin(__builtin_amdgcn_global_load_async_to_lds_b128) && \
    __has_builtin(__builtin_amdgcn_s_wait_asynccnt)
  {
    typedef int v4i_a __attribute__((ext_vector_type(4)));
    typedef __attribute__((address_space(1))) void       gvoid;
    typedef __attribute__((address_space(3))) void       lvoid;
    typedef __attribute__((address_space(1))) v4i_a      gv4i;
    typedef __attribute__((address_space(3))) v4i_a      lv4i;
    for (int i = tid; i < NPAD * KPAD / 8; i += 128) {   // 1600 x 16B transfers
      __builtin_amdgcn_global_load_async_to_lds_b128(
          (gv4i*)(gvoid*)(Wbig + i * 8), (lv4i*)(lvoid*)(Wl + i * 8), 0, 0);
    }
    __builtin_amdgcn_s_wait_asynccnt(0);
  }
#else
  {
    const uint4* wsrc = (const uint4*)Wbig;
    uint4* wdst = (uint4*)Wl;
    for (int i = tid; i < NPAD * KPAD / 8; i += 128) wdst[i] = wsrc[i];
  }
#endif

  const int ebase = blockIdx.x * 64 + wid * 16;  // 16 edges per wave
  const int e  = lane & 15;
  const int p  = lane >> 4;
  const int ge = ebase + e;

  // ---- gather phase: build X tile in bf16 (packed uint2 LDS stores) ----
  if (ge < E) {
    if (p == 0) {
      int s = Xn[ge] - 1;
      Se[wid][e] = s;
      Ne[wid][e] = Xe[ge];                       // RAW 1-indexed scatter id
      Sc[wid][e] = (MUF / (float)SDIM) / dg[ge];
      const float4* fr = (const float4*)(feat + (long)s * LNF);
      __builtin_prefetch(fr, 0, 0);              // global_prefetch_b8
#pragma unroll
      for (int c = 0; c < 16; ++c) {
        float4 f = fr[c];
        uint2 t; t.x = pack2bf(f.x, f.y); t.y = pack2bf(f.z, f.w);
        *(uint2*)&Xl[wid][e * KPAD + c * 4] = t;
      }
      // one-hot + zero pad (k = 128..159): 4x16B zero stores + one b16 poke
      uint4 z; z.x = z.y = z.z = z.w = 0u;
#pragma unroll
      for (int c = 0; c < 4; ++c)
        *(uint4*)&Xl[wid][e * KPAD + 128 + c * 8] = z;
      int et = Et[ge] - 1;
      Xl[wid][e * KPAD + 128 + et] = f2bf(1.0f);
    } else {
      int d = Xe[ge] - 1;
      const float4* fr = (const float4*)(feat + (long)d * LNF);
      __builtin_prefetch(fr, 0, 0);
#pragma unroll
      for (int c = 0; c < 16; ++c) {
        float4 f = fr[c];
        uint2 t; t.x = pack2bf(f.x, f.y); t.y = pack2bf(f.z, f.w);
        *(uint2*)&Xl[wid][e * KPAD + 64 + c * 4] = t;
      }
    }
  }
  __syncthreads();

  // ---- WMMA phase: [16,160]x[160,80] per wave, 25 v_wmma_f32_16x16x32_bf16 ----
  const int m    = lane & 15;                    // A row (edge) / B col (n)
  const int half = lane >> 4;
  const unsigned short* xrow = &Xl[wid][m * KPAD];

#pragma unroll
  for (int nt = 0; nt < 5; ++nt) {
    v8f acc = {0.f, 0.f, 0.f, 0.f, 0.f, 0.f, 0.f, 0.f};
    const unsigned short* wrow = &Wl[(nt * 16 + m) * KPAD];
#pragma unroll
    for (int kc = 0; kc < 5; ++kc) {
      frag_u av, bv;
      // A (16-bit 16x32): v0-3 -> K=half*8..+7 ; v4-7 -> K=16+half*8..+7
      av.q[0] = *(const uint4*)(xrow + kc * 32 + half * 8);
      av.q[1] = *(const uint4*)(xrow + kc * 32 + half * 8 + 16);
      // B (32x16): v0-3 -> K=half*8..+7 ; v4-7 -> K=16+half*8..+7 (matches A)
      bv.q[0] = *(const uint4*)(wrow + kc * 32 + half * 8);
      bv.q[1] = *(const uint4*)(wrow + kc * 32 + half * 8 + 16);
      acc = __builtin_amdgcn_wmma_f32_16x16x32_bf16(false, av.v, false, bv.v,
                                                    (short)0, acc, false, false);
    }
    // D layout: lane n = lane%16, VGPR r -> edge row m = r + 8*half
    if (nt < 4) {
      int col = nt * 16 + m;
      float bb = bxi[col];
#pragma unroll
      for (int r = 0; r < 8; ++r) {
        int row = r + 8 * half;
        Al[wid][row * 64 + col] = DTANH(acc[r] + bb) * Sc[wid][row];
      }
    } else if (m < 8) {
      float bb = brou[m];
#pragma unroll
      for (int r = 0; r < 8; ++r) {
        int row = r + 8 * half;
        Bl[wid][row * 8 + m] = DTANH(acc[r] + bb);
      }
    }
  }
  __syncthreads();

  // ---- per-edge 8x8 matvec + scatter (2 lanes per edge, 4 outputs each) ----
  {
    const int e2 = lane >> 1;
    const int ib = (lane & 1) * 4;
    const bool ok = (ebase + e2) < E;
    int s = ok ? Se[wid][e2] : 0;
    const float* hr = Hc + (long)s * SDIM;
    float he[8];
#pragma unroll
    for (int j = 0; j < 8; ++j) he[j] = hr[j];
    int d = ok ? Ne[wid][e2] : V;                // raw 1-indexed
    float out[4];
#pragma unroll
    for (int ii = 0; ii < 4; ++ii) {
      int i = ib + ii;
      float a = Bl[wid][e2 * 8 + i];
#pragma unroll
      for (int j = 0; j < 8; ++j) a += Al[wid][e2 * 64 + i * 8 + j] * he[j];
      out[ii] = a;
    }
    if (ok && d < V) {                           // id==V dropped (out of range)
#pragma unroll
      for (int ii = 0; ii < 4; ++ii)
        atomicAdd(Hn + (long)d * SDIM + ib + ii, out[ii]);
    }
  }
}

// ---------- softmax-attention readout ----------
__global__ void k_logit_max(const float* __restrict__ H, const float* __restrict__ w1,
                            const float* __restrict__ b1, float* __restrict__ pmax, int V) {
  __shared__ float sm[256];
  const int tid = threadIdx.x;
  float w[8];
#pragma unroll
  for (int j = 0; j < 8; ++j) w[j] = w1[j];
  float b = b1[0], mx = -3.0e38f;
  for (int v = blockIdx.x * 256 + tid; v < V; v += gridDim.x * 256) {
    const float* h = H + (long)v * SDIM;
    float l = b;
#pragma unroll
    for (int j = 0; j < 8; ++j) l += h[j] * w[j];
    mx = fmaxf(mx, l);
  }
  sm[tid] = mx; __syncthreads();
  for (int k = 128; k > 0; k >>= 1) {
    if (tid < k) sm[tid] = fmaxf(sm[tid], sm[tid + k]);
    __syncthreads();
  }
  if (tid == 0) pmax[blockIdx.x] = sm[0];
}

__global__ void k_maxred(const float* __restrict__ pmax, float* __restrict__ gmax) {
  __shared__ float sm[256];
  const int tid = threadIdx.x;
  sm[tid] = pmax[tid]; __syncthreads();
  for (int k = 128; k > 0; k >>= 1) {
    if (tid < k) sm[tid] = fmaxf(sm[tid], sm[tid + k]);
    __syncthreads();
  }
  if (tid == 0) gmax[0] = sm[0];
}

__global__ void k_expsum(const float* __restrict__ H, const float* __restrict__ w1,
                         const float* __restrict__ b1, const float* __restrict__ gmax,
                         float* __restrict__ psum, int V) {
  __shared__ float sm[256];
  const int tid = threadIdx.x;
  float w[8];
#pragma unroll
  for (int j = 0; j < 8; ++j) w[j] = w1[j];
  float b = b1[0], mx = gmax[0];
  float acc[9];
#pragma unroll
  for (int s = 0; s < 9; ++s) acc[s] = 0.0f;
  for (int v = blockIdx.x * 256 + tid; v < V; v += gridDim.x * 256) {
    const float* h = H + (long)v * SDIM;
    float l = b;
#pragma unroll
    for (int j = 0; j < 8; ++j) l += h[j] * w[j];
    float e = expf(l - mx);
    acc[0] += e;
#pragma unroll
    for (int j = 0; j < 8; ++j) acc[1 + j] += e * h[j];
  }
  for (int s = 0; s < 9; ++s) {
    sm[tid] = acc[s]; __syncthreads();
    for (int k = 128; k > 0; k >>= 1) {
      if (tid < k) sm[tid] += sm[tid + k];
      __syncthreads();
    }
    if (tid == 0) psum[blockIdx.x * 9 + s] = sm[0];
    __syncthreads();
  }
}

__global__ void k_final(const float* __restrict__ psum, float* __restrict__ out) {
  __shared__ float sm[256];
  const int tid = threadIdx.x;
  float acc[9], res[9];
#pragma unroll
  for (int s = 0; s < 9; ++s) acc[s] = psum[tid * 9 + s];
  for (int s = 0; s < 9; ++s) {
    sm[tid] = acc[s]; __syncthreads();
    for (int k = 128; k > 0; k >>= 1) {
      if (tid < k) sm[tid] += sm[tid + k];
      __syncthreads();
    }
    res[s] = sm[0]; __syncthreads();
  }
  if (tid == 0) {
    float denom = res[0];
#pragma unroll
    for (int s = 0; s < 8; ++s) out[s] = tanhf(res[1 + s] / denom);
  }
}

extern "C" void kernel_launch(void* const* d_in, const int* in_sizes, int n_in,
                              void* d_out, int out_size, void* d_ws, size_t ws_size,
                              hipStream_t stream) {
  const float* feat  = (const float*)d_in[0];
  const int*   Xn    = (const int*)d_in[1];
  const int*   Xe    = (const int*)d_in[2];
  const int*   Et    = (const int*)d_in[3];
  const float* dg    = (const float*)d_in[4];
  const float* Hinit = (const float*)d_in[5];
  const float* Wxi   = (const float*)d_in[6];
  const float* bxi   = (const float*)d_in[7];
  const float* Wrou  = (const float*)d_in[8];
  const float* brou  = (const float*)d_in[9];
  const float* W1    = (const float*)d_in[10];
  const float* b1    = (const float*)d_in[11];
  float* out = (float*)d_out;

  const int V = in_sizes[0] / LNF;
  const int E = in_sizes[1];

  char* ws = (char*)d_ws;
  unsigned short* Wbig = (unsigned short*)ws;                       // 25600 B
  float* Ha   = (float*)(ws + 32768);
  float* Hb   = (float*)(ws + 32768 + (size_t)V * SDIM * 4);
  float* pmax = (float*)(ws + 32768 + (size_t)V * SDIM * 8);
  float* gmax = pmax + 256;
  float* psum = gmax + 64;

  k_prep<<<dim3(1024), dim3(256), 0, stream>>>(Wxi, Wrou, Hinit, Wbig, Ha, V);

  float* cur = Ha;
  float* nxt = Hb;
  for (int t = 0; t < TSTEP; ++t) {
    k_zero<<<dim3(1024), dim3(256), 0, stream>>>(nxt, V * SDIM);
    k_step<<<dim3((E + 63) / 64), dim3(128), 0, stream>>>(
        feat, Xn, Xe, Et, dg, Wbig, bxi, brou, cur, nxt, E, V);
    float* tmp = cur; cur = nxt; nxt = tmp;
  }

  k_logit_max<<<dim3(256), dim3(256), 0, stream>>>(cur, W1, b1, pmax, V);
  k_maxred  <<<dim3(1),   dim3(256), 0, stream>>>(pmax, gmax);
  k_expsum  <<<dim3(256), dim3(256), 0, stream>>>(cur, W1, b1, gmax, psum, V);
  k_final   <<<dim3(1),   dim3(256), 0, stream>>>(psum, out);
}
